// CustomWindowMHA_67594195304486
// MI455X (gfx1250) — matrix operand order repaired and checked by reading, not compiled, not run on hardware
//
#include <hip/hip_runtime.h>

// ---------------- problem constants ----------------
#define S_LEN   2048
#define BATCH   2
#define DMODEL  1024
#define NHEAD   16
#define DHEAD   64
#define WINDOW  128
#define DIL     4

typedef __attribute__((ext_vector_type(16))) __bf16 v16bf;
typedef __attribute__((ext_vector_type(8)))  float  v8f;
typedef int v4i_gcc __attribute__((vector_size(4 * sizeof(int))));

struct __align__(16) B128 { unsigned int x[4]; };
union V16 { v16bf v; B128 q[2]; unsigned short s[16]; };

__device__ __forceinline__ unsigned short f2bf(float f) {
  unsigned int u = __float_as_uint(f);
  u += 0x7FFFu + ((u >> 16) & 1u);          // round-to-nearest-even
  return (unsigned short)(u >> 16);
}

__device__ __forceinline__ v8f bf_wmma(const V16& a, const V16& b, v8f c) {
  return __builtin_amdgcn_wmma_f32_16x16x32_bf16(false, a.v, false, b.v,
                                                 (short)0, c, false, false);
}

// ---- CDNA5 async global->LDS copy (ASYNCcnt domain), builtin or asm ----
__device__ __forceinline__ void async_cp_b128(unsigned short* lptr,
                                              const unsigned short* gptr) {
#if defined(__has_builtin) && __has_builtin(__builtin_amdgcn_global_load_async_to_lds_b128)
  __builtin_amdgcn_global_load_async_to_lds_b128(
      (__attribute__((address_space(1))) v4i_gcc*)gptr,
      (__attribute__((address_space(3))) v4i_gcc*)lptr, 0, 0);
#else
  asm volatile("global_load_async_to_lds_b128 %0, %1, off"
               :
               : "v"((unsigned int)(unsigned long long)lptr), "v"(gptr)
               : "memory");
#endif
}

__device__ __forceinline__ void wait_async0() {
#if defined(__has_builtin) && __has_builtin(__builtin_amdgcn_s_wait_asynccnt)
  __builtin_amdgcn_s_wait_asynccnt(0);
#else
  asm volatile("s_wait_asynccnt 0x0" ::: "memory");
#endif
}

// ---------------- fp32 -> bf16 convert ----------------
__global__ void cvt_bf16_kernel(const float* __restrict__ src,
                                unsigned short* __restrict__ dst, int n) {
  int i = blockIdx.x * blockDim.x + threadIdx.x;
  int stride = gridDim.x * blockDim.x;
  for (; i < n; i += stride) dst[i] = f2bf(src[i]);
}

// ---------------- unified GEMM with async-LDS B staging ----------------
// C(m,n) = sum_k A[m][k] * Bw[n][k].  Block = 256 thr (8 waves) computing
// 32 rows x 256 cols; B K-slices (256 x 64) double-buffered in LDS via
// GLOBAL_LOAD_ASYNC_TO_LDS_B128, consumed through ds_load_b128.
// MODE 0: QKV scatter epilogue.  MODE 1: plain f32 output.
#define KSLICE 64

template <int MODE>
__global__ void gemm_kernel(const unsigned short* __restrict__ Ab,
                            const unsigned short* __restrict__ Bw,
                            unsigned short* __restrict__ qbuf,
                            unsigned short* __restrict__ kbuf,
                            unsigned short* __restrict__ vtb,
                            unsigned short* __restrict__ vtd,
                            float* __restrict__ outf,
                            int numN) {
  __shared__ __align__(16) unsigned short ldsB[2][256 * KSLICE];   // 64 KB

  int mt = blockIdx.x / numN;
  int nt = blockIdx.x % numN;
  int tid  = threadIdx.x;
  int w    = tid >> 5;
  int lane = tid & 31;
  int half = lane >> 4, l16 = lane & 15;

  int m0      = mt * 32 + (w & 1) * 16;
  int n0block = nt * 256;
  int n0      = n0block + (w >> 1) * 64;

  const unsigned short* arow = Ab + (size_t)(m0 + l16) * DMODEL;
  int ac0 = half * 16;

  // async fill of one 256x64 B slice: 8 x b128 per thread
  auto fill = [&](int buf, int k) {
#pragma unroll
    for (int i = 0; i < 8; ++i) {
      int e   = i * 256 + tid;
      int row = e >> 3, ch = e & 7;
      const unsigned short* g = Bw + (size_t)(n0block + row) * DMODEL + k + ch * 8;
      async_cp_b128(&ldsB[buf][row * KSLICE + ch * 8], g);
    }
  };

  fill(0, 0);
  v8f acc0 = {}, acc1 = {}, acc2 = {}, acc3 = {};
  wait_async0();
  __syncthreads();

  for (int k = 0; k < DMODEL; k += KSLICE) {
    int buf = (k >> 6) & 1;
    if (k + KSLICE < DMODEL) fill(buf ^ 1, k + KSLICE);

    // A operands for both 32-wide k-steps of this slice (direct global)
    V16 a0, a1;
    a0.q[0] = *reinterpret_cast<const B128*>(arow + k + ac0);
    a0.q[1] = *reinterpret_cast<const B128*>(arow + k + ac0 + 8);
    a1.q[0] = *reinterpret_cast<const B128*>(arow + k + 32 + ac0);
    a1.q[1] = *reinterpret_cast<const B128*>(arow + k + 32 + ac0 + 8);

    const unsigned short* bl =
        &ldsB[buf][((w >> 1) * 64 + l16) * KSLICE + half * 16];
#pragma unroll
    for (int ks = 0; ks < 2; ++ks) {
      const unsigned short* blk = bl + ks * 32;
      V16 b0, b1, b2, b3;
      b0.q[0] = *reinterpret_cast<const B128*>(blk);
      b0.q[1] = *reinterpret_cast<const B128*>(blk + 8);
      b1.q[0] = *reinterpret_cast<const B128*>(blk + 16 * KSLICE);
      b1.q[1] = *reinterpret_cast<const B128*>(blk + 16 * KSLICE + 8);
      b2.q[0] = *reinterpret_cast<const B128*>(blk + 32 * KSLICE);
      b2.q[1] = *reinterpret_cast<const B128*>(blk + 32 * KSLICE + 8);
      b3.q[0] = *reinterpret_cast<const B128*>(blk + 48 * KSLICE);
      b3.q[1] = *reinterpret_cast<const B128*>(blk + 48 * KSLICE + 8);
      const V16& a = ks ? a1 : a0;
      acc0 = bf_wmma(a, b0, acc0);
      acc1 = bf_wmma(a, b1, acc1);
      acc2 = bf_wmma(a, b2, acc2);
      acc3 = bf_wmma(a, b3, acc3);
    }

    wait_async0();        // our fills for next slice landed
    __syncthreads();      // everyone done reading cur / filling next
  }

  v8f accs[4] = {acc0, acc1, acc2, acc3};
#pragma unroll
  for (int t = 0; t < 4; ++t) {
#pragma unroll
    for (int r = 0; r < 8; ++r) {
      int m = m0 + r + 8 * half;
      int n = n0 + 16 * t + l16;
      if (MODE == 1) {
        outf[(size_t)m * DMODEL + n] = accs[t][r];
      } else {
        int bb = m >> 11, s = m & (S_LEN - 1);
        int sec = n >> 10, nn = n & (DMODEL - 1);
        int h = nn >> 6, d = nn & (DHEAD - 1);
        unsigned short bv = f2bf(accs[t][r]);
        int bh = bb * NHEAD + h;
        if (sec == 0) {
          qbuf[(size_t)(bh * S_LEN + s) * DHEAD + d] = bv;
        } else if (sec == 1) {
          kbuf[(size_t)(bh * S_LEN + s) * DHEAD + d] = bv;
        } else {
          vtb[(size_t)(bh * DHEAD + d) * S_LEN + s] = bv;
          vtd[(size_t)((bh * 4 + (s & 3)) * DHEAD + d) * (S_LEN / 4) + (s >> 2)] = bv;
        }
      }
    }
  }
}

// ---------------- sparse-window flash attention ----------------
// Block = 4 waves; wave c (=residue class) owns 16 query rows {i064 + c + 4t}.
// Dilated phase: gathered keys j = 4*jj + c processed densely (near-100% tile
// utilization). Window phase: dense keys, window-only mask (no double count).
// Row-sum of P computed as a 5th WMMA against an all-ones B tile (kills the
// ds_bpermute sum tree); only row-max needs a shuffle reduction.

template <bool DILATED>
__device__ __forceinline__ void attn_block(
    const V16& aq0, const V16& aq1, const V16& ones,
    v8f& O0, v8f& O1, v8f& O2, v8f& O3, v8f& Ls,
    float (&m_)[8],
    unsigned short* pl,
    const unsigned short* Kh,
    const unsigned short* Vbase, int vstride,
    int kb2, int klast, int K0, int c, int ibase,
    int half, int l16, int c0) {
  float sc[2][8];
#pragma unroll
  for (int t2 = 0; t2 < 2; ++t2) {
    int kt = kb2 + 16 * t2;
    if (kt <= klast) {
      int kidx = kt + l16;
      int key  = DILATED ? (4 * kidx + c) : kidx;
      const unsigned short* krow = Kh + (size_t)key * DHEAD + half * 16;
      V16 b0, b1;
      b0.q[0] = *reinterpret_cast<const B128*>(krow);
      b0.q[1] = *reinterpret_cast<const B128*>(krow + 8);
      b1.q[0] = *reinterpret_cast<const B128*>(krow + 32);
      b1.q[1] = *reinterpret_cast<const B128*>(krow + 40);
      v8f s = {};
      s = bf_wmma(aq0, b0, s);
      s = bf_wmma(aq1, b1, s);
#pragma unroll
      for (int r = 0; r < 8; ++r) {
        int t   = r + 8 * half;
        int row = ibase + 4 * t;
        bool ok;
        if (DILATED) ok = (kidx <= t + K0);
        else         ok = (kidx <= row) && (kidx >= row - (WINDOW - 1));
        sc[t2][r] = ok ? s[r] * 0.125f : -1e30f;
      }
    } else {
#pragma unroll
      for (int r = 0; r < 8; ++r) sc[t2][r] = -1e30f;
    }
  }

  // online softmax: row-max shuffle tree, exp, P -> LDS (bf16)
  float alpha[8];
#pragma unroll
  for (int r = 0; r < 8; ++r) {
    float mb = fmaxf(sc[0][r], sc[1][r]);
    mb = fmaxf(mb, __shfl_xor(mb, 1, 32));
    mb = fmaxf(mb, __shfl_xor(mb, 2, 32));
    mb = fmaxf(mb, __shfl_xor(mb, 4, 32));
    mb = fmaxf(mb, __shfl_xor(mb, 8, 32));
    float mn = fmaxf(m_[r], mb);
    alpha[r] = __expf(m_[r] - mn);
    float p0 = __expf(sc[0][r] - mn);
    float p1 = __expf(sc[1][r] - mn);
    pl[(r + 8 * half) * 32 + l16]      = f2bf(p0);
    pl[(r + 8 * half) * 32 + 16 + l16] = f2bf(p1);
    m_[r] = mn;
  }
#pragma unroll
  for (int r = 0; r < 8; ++r) {
    O0[r] *= alpha[r]; O1[r] *= alpha[r];
    O2[r] *= alpha[r]; O3[r] *= alpha[r];
    Ls[r] *= alpha[r];
  }

  // P tile back as A operand
  V16 pa;
  pa.q[0] = *reinterpret_cast<const B128*>(pl + l16 * 32 + c0);
  pa.q[1] = *reinterpret_cast<const B128*>(pl + l16 * 32 + c0 + 8);

  // V tiles late: minimal live range
  const unsigned short* vr0 = Vbase + (size_t)( 0 + l16) * vstride + kb2 + half * 16;
  const unsigned short* vr1 = Vbase + (size_t)(16 + l16) * vstride + kb2 + half * 16;
  const unsigned short* vr2 = Vbase + (size_t)(32 + l16) * vstride + kb2 + half * 16;
  const unsigned short* vr3 = Vbase + (size_t)(48 + l16) * vstride + kb2 + half * 16;
  V16 bv;
  bv.q[0] = *reinterpret_cast<const B128*>(vr0);
  bv.q[1] = *reinterpret_cast<const B128*>(vr0 + 8);
  O0 = bf_wmma(pa, bv, O0);
  bv.q[0] = *reinterpret_cast<const B128*>(vr1);
  bv.q[1] = *reinterpret_cast<const B128*>(vr1 + 8);
  O1 = bf_wmma(pa, bv, O1);
  bv.q[0] = *reinterpret_cast<const B128*>(vr2);
  bv.q[1] = *reinterpret_cast<const B128*>(vr2 + 8);
  O2 = bf_wmma(pa, bv, O2);
  bv.q[0] = *reinterpret_cast<const B128*>(vr3);
  bv.q[1] = *reinterpret_cast<const B128*>(vr3 + 8);
  O3 = bf_wmma(pa, bv, O3);
  // row-sum accumulator: every column of (P x ones) is sum_k P[row][k]
  Ls = bf_wmma(pa, ones, Ls);
}

__global__ void __launch_bounds__(128, 1)
attn_kernel(const unsigned short* __restrict__ qbuf,
            const unsigned short* __restrict__ kbuf,
            const unsigned short* __restrict__ vtb,
            const unsigned short* __restrict__ vtd,
            unsigned short* __restrict__ attnb) {
  __shared__ __align__(16) unsigned short plds[4][16 * 32];

  int bh   = blockIdx.x >> 5;          // 32 blocks of 64 rows per (b,h)
  int blk  = blockIdx.x & 31;
  int b = bh >> 4, h = bh & 15;
  int w    = threadIdx.x >> 5;         // wave == residue class c
  int lane = threadIdx.x & 31;
  int half = lane >> 4, l16 = lane & 15;
  int c0 = half * 16;

  int i064  = blk * 64;
  int c     = w;
  int ibase = i064 + c;                // rows: ibase + 4t, t = 0..15

  const unsigned short* Qh  = qbuf + (size_t)bh * S_LEN * DHEAD;
  const unsigned short* Kh  = kbuf + (size_t)bh * S_LEN * DHEAD;
  const unsigned short* Vth = vtb  + (size_t)bh * DHEAD * S_LEN;
  const unsigned short* Vtd = vtd  + (size_t)(bh * 4 + c) * DHEAD * (S_LEN / 4);
  unsigned short* pl = plds[w];

  // all-ones bf16 B tile for the row-sum WMMA
  V16 ones;
#pragma unroll
  for (int i = 0; i < 16; ++i) ones.s[i] = 0x3F80u;   // bf16 1.0

  // Q tile, A-operand layout (rows strided by 4)
  const unsigned short* qrow = Qh + (size_t)(ibase + 4 * l16) * DHEAD;
  V16 aq0, aq1;
  aq0.q[0] = *reinterpret_cast<const B128*>(qrow + c0);
  aq0.q[1] = *reinterpret_cast<const B128*>(qrow + c0 + 8);
  aq1.q[0] = *reinterpret_cast<const B128*>(qrow + 32 + c0);
  aq1.q[1] = *reinterpret_cast<const B128*>(qrow + 32 + c0 + 8);

  float m_[8];
  v8f O0 = {}, O1 = {}, O2 = {}, O3 = {}, Ls = {};
#pragma unroll
  for (int r = 0; r < 8; ++r) m_[r] = -1e30f;

  // ---- dilated phase: gathered keys j = 4*jj + c, mask jj <= t + K0 ----
  int K0 = (i064 - (WINDOW + DIL)) / 4;          // exact (both mult of 4)
  int klast_d = K0 + 15;
  for (int kb2 = 0; kb2 <= klast_d; kb2 += 32)
    attn_block<true>(aq0, aq1, ones, O0, O1, O2, O3, Ls, m_, pl,
                     Kh, Vtd, S_LEN / 4, kb2, klast_d, K0, c, ibase,
                     half, l16, c0);

  // ---- window phase: dense keys, window-only mask ----
  int w0 = ibase - (WINDOW - 1);
  if (w0 < 0) w0 = 0;
  w0 &= ~7;                                      // 16B-aligned V loads
  int klast_w = ibase + 60;                      // largest row of this wave
  for (int kb2 = w0; kb2 <= klast_w; kb2 += 32)
    attn_block<false>(aq0, aq1, ones, O0, O1, O2, O3, Ls, m_, pl,
                      Kh, Vth, S_LEN, kb2, klast_w, K0, c, ibase,
                      half, l16, c0);

  // epilogue: normalize, write bf16 attention output (B*S, D) row-major
  v8f Os[4] = {O0, O1, O2, O3};
#pragma unroll
  for (int t = 0; t < 4; ++t) {
#pragma unroll
    for (int r = 0; r < 8; ++r) {
      int row = ibase + 4 * (r + 8 * half);
      float v = Os[t][r] / Ls[r];
      attnb[(size_t)(b * S_LEN + row) * DMODEL + h * DHEAD + 16 * t + l16] = f2bf(v);
    }
  }
}

// ---------------- host launcher ----------------
extern "C" void kernel_launch(void* const* d_in, const int* in_sizes, int n_in,
                              void* d_out, int out_size, void* d_ws, size_t ws_size,
                              hipStream_t stream) {
  (void)in_sizes; (void)n_in; (void)out_size; (void)ws_size;

  const float* x   = (const float*)d_in[0];   // (2, 2048, 1024)
  const float* qkv = (const float*)d_in[1];   // (3072, 1024)
  const float* wo  = (const float*)d_in[2];   // (1024, 1024)
  float* out = (float*)d_out;                 // (2, 2048, 1024) f32

  const size_t MROWS  = (size_t)BATCH * S_LEN;                  // 4096
  const size_t SZ_X   = MROWS * DMODEL;                         // 4 Mi
  const size_t SZ_QKV = (size_t)3 * DMODEL * DMODEL;            // 3 Mi
  const size_t SZ_WO  = (size_t)DMODEL * DMODEL;                // 1 Mi
  const size_t SZ_HDS = (size_t)BATCH * NHEAD * S_LEN * DHEAD;  // 4 Mi

  unsigned short* xb    = (unsigned short*)d_ws;
  unsigned short* qkvb  = xb    + SZ_X;
  unsigned short* wob   = qkvb  + SZ_QKV;
  unsigned short* qbuf  = wob   + SZ_WO;
  unsigned short* kbuf  = qbuf  + SZ_HDS;
  unsigned short* vtb   = kbuf  + SZ_HDS;
  unsigned short* vtd   = vtb   + SZ_HDS;     // residue-compacted V^T
  unsigned short* attnb = vtd   + SZ_HDS;

  cvt_bf16_kernel<<<1024, 256, 0, stream>>>(x,   xb,   (int)SZ_X);
  cvt_bf16_kernel<<<1024, 256, 0, stream>>>(qkv, qkvb, (int)SZ_QKV);
  cvt_bf16_kernel<<<1024, 256, 0, stream>>>(wo,  wob,  (int)SZ_WO);

  // QKV GEMM: M=4096, N=3072 -> (4096/32)*(3072/256) = 1536 blocks
  gemm_kernel<0><<<1536, 256, 0, stream>>>(xb, qkvb, qbuf, kbuf, vtb, vtd,
                                           nullptr, 12);

  // Attention: B*H*(S/64) = 1024 blocks of 128 threads (4 waves = 4 residues)
  attn_kernel<<<1024, 128, 0, stream>>>(qbuf, kbuf, vtb, vtd, attnb);

  // Output GEMM: M=4096, N=1024 -> (4096/32)*(1024/256) = 512 blocks
  gemm_kernel<1><<<512, 256, 0, stream>>>(attnb, wob, nullptr, nullptr, nullptr,
                                          nullptr, out, 4);
}
